// MambaSequenceClassifier_23106924053243
// MI455X (gfx1250) — compile-verified, matrix operand order checked by reading, not run on hardware
//
#include <hip/hip_runtime.h>
#include <hip/hip_bf16.h>
#include <stdint.h>

// ---------------------------------------------------------------------------
// Mamba sequence classifier, MI455X (gfx1250), wave32 + WMMA bf16 path.
// GEMMs: v_wmma_f32_16x16x32_bf16, branch-free k-loop, register
// double-buffered (software pipelined) fragment loads so WMMA issue
// overlaps global memory latency instead of draining loadcnt each tile.
// ---------------------------------------------------------------------------

#define BB   16
#define LL   100
#define DIN  6144
#define HH   1024
#define DI   2048
#define NN   16
#define DTR  64
#define KK   4
#define NC   16
#define BL   (BB * LL)   // 1600

typedef __attribute__((ext_vector_type(16))) __bf16       v16bf;
typedef __attribute__((ext_vector_type(8)))  float        v8f;
typedef __attribute__((ext_vector_type(4)))  unsigned int v4u;

union FragBF {
    v4u   u4[2];
    v16bf v;
};

// Two contiguous b128 loads per fragment (ISA 16-bit fragment layout:
// per-lane K = phase+0..7 and phase+16..23).
#define LOAD_FRAG(fr, ptr)                                          \
    do {                                                            \
        (fr).u4[0] = *reinterpret_cast<const v4u*>(ptr);            \
        (fr).u4[1] = *reinterpret_cast<const v4u*>((ptr) + 16);     \
    } while (0)

static __device__ __forceinline__ unsigned short f32_to_bf16(float f) {
    unsigned int u = __float_as_uint(f);
    unsigned int lsb = (u >> 16) & 1u;
    u += 0x7fffu + lsb;             // round-to-nearest-even
    return (unsigned short)(u >> 16);
}

// -------------------- elementwise f32 -> bf16 --------------------
__global__ void k_cvt_bf16(const float* __restrict__ in,
                           unsigned short* __restrict__ out, int n) {
    int i = blockIdx.x * blockDim.x + threadIdx.x;
    if (i < n) out[i] = f32_to_bf16(in[i]);
}

// -------------------- W (R x C) f32 -> Wt (C x R) bf16 --------------------
__global__ void k_transpose_cvt(const float* __restrict__ in,
                                unsigned short* __restrict__ out,
                                int R, int C) {
    int i = blockIdx.x * blockDim.x + threadIdx.x;
    if (i >= R * C) return;
    int r = i / C, c = i % C;
    out[(size_t)c * R + r] = f32_to_bf16(in[i]);
}

// -------------------- WMMA bf16 GEMM --------------------
// C[M,N] = A[M,K](bf16) * Wt[N,K](bf16)^T + bias
// grid = (N/(NT*16), ceil(M/128)), block = 256 (8 waves).
// Wave w: rows [(blockIdx.y*8+w)*16, +16), cols [blockIdx.x*NT*16, +NT*16).
// Requires: N % (NT*16) == 0, K % 64 == 0 (pipelined in pairs of 32-steps).
template <int NT>
__global__ void __launch_bounds__(256)
k_gemm_bf16(int M, int Ntot, int Ktot,
            const unsigned short* __restrict__ A,
            const unsigned short* __restrict__ Wt,
            const float* __restrict__ bias,
            float* __restrict__ Cf,            // optional f32 out
            unsigned short* __restrict__ Cb) { // optional bf16 out
    const int lane = threadIdx.x & 31;
    const int wid  = threadIdx.x >> 5;
    const int m0   = (blockIdx.y * 8 + wid) * 16;
    if (m0 >= M) return;                       // wave-uniform: EXEC stays full
    const int n_base = blockIdx.x * (NT * 16);

    const int l15    = lane & 15;
    const int kphase = (lane >> 4) << 3;       // 0 for lanes 0-15, 8 for 16-31

    v8f acc[NT];
#pragma unroll
    for (int t = 0; t < NT; ++t) acc[t] = (v8f){0,0,0,0,0,0,0,0};

    const unsigned short* pa = A + (size_t)(m0 + l15) * Ktot + kphase;
    const unsigned short* pb[NT];
#pragma unroll
    for (int t = 0; t < NT; ++t)
        pb[t] = Wt + (size_t)(n_base + t * 16 + l15) * Ktot + kphase;

    FragBF fa0, fa1, fb0[NT], fb1[NT];

    // prologue: buffer 0 <- k-step 0
    LOAD_FRAG(fa0, pa);
#pragma unroll
    for (int t = 0; t < NT; ++t) LOAD_FRAG(fb0[t], pb[t]);

    int kk = 0;
    for (; kk < Ktot - 64; kk += 64) {
        // buffer 1 <- k-step kk+32 (in flight during buffer-0 WMMAs)
        LOAD_FRAG(fa1, pa + kk + 32);
#pragma unroll
        for (int t = 0; t < NT; ++t) LOAD_FRAG(fb1[t], pb[t] + kk + 32);
#pragma unroll
        for (int t = 0; t < NT; ++t)
            acc[t] = __builtin_amdgcn_wmma_f32_16x16x32_bf16(
                false, fa0.v, false, fb0[t].v, (short)0, acc[t], false, false);
        // buffer 0 <- k-step kk+64 (in flight during buffer-1 WMMAs)
        LOAD_FRAG(fa0, pa + kk + 64);
#pragma unroll
        for (int t = 0; t < NT; ++t) LOAD_FRAG(fb0[t], pb[t] + kk + 64);
#pragma unroll
        for (int t = 0; t < NT; ++t)
            acc[t] = __builtin_amdgcn_wmma_f32_16x16x32_bf16(
                false, fa1.v, false, fb1[t].v, (short)0, acc[t], false, false);
    }
    // epilogue: kk == Ktot-64; last two k-steps, no further prefetch
    LOAD_FRAG(fa1, pa + kk + 32);
#pragma unroll
    for (int t = 0; t < NT; ++t) LOAD_FRAG(fb1[t], pb[t] + kk + 32);
#pragma unroll
    for (int t = 0; t < NT; ++t)
        acc[t] = __builtin_amdgcn_wmma_f32_16x16x32_bf16(
            false, fa0.v, false, fb0[t].v, (short)0, acc[t], false, false);
#pragma unroll
    for (int t = 0; t < NT; ++t)
        acc[t] = __builtin_amdgcn_wmma_f32_16x16x32_bf16(
            false, fa1.v, false, fb1[t].v, (short)0, acc[t], false, false);

    const int mbase = m0 + ((lane >> 4) << 3);  // D rows: r + (lane>=16 ? 8 : 0)
#pragma unroll
    for (int t = 0; t < NT; ++t) {
        const int n = n_base + t * 16 + l15;
        const float bv = bias ? bias[n] : 0.0f;
#pragma unroll
        for (int r = 0; r < 8; ++r) {
            float v = acc[t][r] + bv;
            size_t off = (size_t)(mbase + r) * Ntot + n;
            if (Cf) Cf[off] = v;
            if (Cb) Cb[off] = f32_to_bf16(v);
        }
    }
}

// -------------------- depthwise causal conv (K=4) + SiLU --------------------
__global__ void k_conv_silu(const float* __restrict__ xz,
                            const float* __restrict__ conv_w,
                            const float* __restrict__ conv_b,
                            float* __restrict__ xc_f,
                            unsigned short* __restrict__ xc_b) {
    int idx = blockIdx.x * blockDim.x + threadIdx.x;   // over BL*DI
    if (idx >= BL * DI) return;
    int d  = idx % DI;
    int bl = idx / DI;
    int l  = bl % LL;
    float s = conv_b[d];
#pragma unroll
    for (int k = 0; k < KK; ++k) {
        int lp = l + k - (KK - 1);
        if (lp >= 0)
            s += conv_w[d * KK + k] * xz[(size_t)(bl + lp - l) * (2 * DI) + d];
    }
    float v = s / (1.0f + expf(-s));                    // SiLU
    xc_f[(size_t)bl * DI + d] = v;
    xc_b[(size_t)bl * DI + d] = f32_to_bf16(v);
}

// -------------------- dt = proj[:, :DTR] -> bf16 --------------------
__global__ void k_slice_dt(const float* __restrict__ proj,
                           unsigned short* __restrict__ dt_b) {
    int i = blockIdx.x * blockDim.x + threadIdx.x;     // over BL*DTR
    if (i >= BL * DTR) return;
    int row = i / DTR, c = i % DTR;
    dt_b[i] = f32_to_bf16(proj[(size_t)row * (DTR + 2 * NN) + c]);
}

// -------------------- selective scan + D skip + SiLU(z) gate --------------------
__global__ void __launch_bounds__(256)
k_scan(const float* __restrict__ delta_raw,  // (BL, DI) pre-softplus
       const float* __restrict__ xc,         // (BL, DI)
       const float* __restrict__ proj,       // (BL, 96): [64:80)=B, [80:96)=C
       const float* __restrict__ xz,         // (BL, 2*DI): z = cols [DI, 2*DI)
       const float* __restrict__ A_log,      // (DI, N)
       const float* __restrict__ Dp,         // (DI)
       unsigned short* __restrict__ y_b) {   // (BL, DI) bf16
    const int d = blockIdx.x * blockDim.x + threadIdx.x;
    const int b = blockIdx.y;
    __shared__ float sBC[2 * NN];

    float hreg[NN], Areg[NN];
#pragma unroll
    for (int n = 0; n < NN; ++n) {
        Areg[n] = -expf(A_log[(size_t)d * NN + n]);
        hreg[n] = 0.0f;
    }
    const float Dd = Dp[d];

    for (int l = 0; l < LL; ++l) {
        const size_t row = (size_t)(b * LL + l);
        __syncthreads();
        if (threadIdx.x < 2 * NN)
            sBC[threadIdx.x] = proj[row * (DTR + 2 * NN) + DTR + threadIdx.x];
        __syncthreads();

        float dr = delta_raw[row * DI + d];
        float delta = (dr > 20.0f) ? dr : log1pf(expf(dr));   // softplus
        float xcv = xc[row * DI + d];
        float dx = delta * xcv;
        float y = 0.0f;
#pragma unroll
        for (int n = 0; n < NN; ++n) {
            hreg[n] = expf(delta * Areg[n]) * hreg[n] + dx * sBC[n];
            y += hreg[n] * sBC[NN + n];
        }
        y += Dd * xcv;
        float z = xz[row * (2 * DI) + DI + d];
        y *= z / (1.0f + expf(-z));                           // * SiLU(z)
        y_b[row * DI + d] = f32_to_bf16(y);
    }
}

// ---------------------------------------------------------------------------
extern "C" void kernel_launch(void* const* d_in, const int* in_sizes, int n_in,
                              void* d_out, int out_size, void* d_ws, size_t ws_size,
                              hipStream_t stream) {
    const float* x         = (const float*)d_in[0];   // (B,L,DIN)
    const float* W_in      = (const float*)d_in[1];   // (DIN,H)
    const float* b_in      = (const float*)d_in[2];   // (H)
    const float* in_proj_w = (const float*)d_in[3];   // (H, 2*DI)
    const float* conv_w    = (const float*)d_in[4];   // (DI,K)
    const float* conv_b    = (const float*)d_in[5];   // (DI)
    const float* x_proj_w  = (const float*)d_in[6];   // (DI, 96)
    const float* dt_proj_w = (const float*)d_in[7];   // (DTR, DI)
    const float* dt_proj_b = (const float*)d_in[8];   // (DI)
    const float* A_log     = (const float*)d_in[9];   // (DI,N)
    const float* Dp        = (const float*)d_in[10];  // (DI)
    const float* out_projw = (const float*)d_in[11];  // (DI,H)
    const float* W_out     = (const float*)d_in[12];  // (H,NC)
    const float* b_out     = (const float*)d_in[13];  // (NC)
    float* logits          = (float*)d_out;           // (B,L,NC) f32

    // ---- workspace bump allocator (256B aligned) ----
    size_t off = 0;
    auto alloc = [&](size_t bytes) -> char* {
        char* p = (char*)d_ws + off;
        off = (off + bytes + 255) & ~(size_t)255;
        return p;
    };
    unsigned short* x_bf    = (unsigned short*)alloc((size_t)BL * DIN * 2);
    unsigned short* Win_t   = (unsigned short*)alloc((size_t)HH * DIN * 2);
    unsigned short* h_bf    = (unsigned short*)alloc((size_t)BL * HH * 2);
    unsigned short* inp_t   = (unsigned short*)alloc((size_t)(2 * DI) * HH * 2);
    float*          xz_f    = (float*)alloc((size_t)BL * 2 * DI * 4);
    float*          xc_f    = (float*)alloc((size_t)BL * DI * 4);
    unsigned short* xc_b    = (unsigned short*)alloc((size_t)BL * DI * 2);
    unsigned short* xproj_t = (unsigned short*)alloc((size_t)(DTR + 2 * NN) * DI * 2);
    float*          proj_f  = (float*)alloc((size_t)BL * (DTR + 2 * NN) * 4);
    unsigned short* dt_bf   = (unsigned short*)alloc((size_t)BL * DTR * 2);
    unsigned short* dtp_t   = (unsigned short*)alloc((size_t)DI * DTR * 2);
    float*          delta_f = (float*)alloc((size_t)BL * DI * 4);
    unsigned short* y_bf    = (unsigned short*)alloc((size_t)BL * DI * 2);
    unsigned short* outp_t  = (unsigned short*)alloc((size_t)HH * DI * 2);
    unsigned short* out_bf  = (unsigned short*)alloc((size_t)BL * HH * 2);
    unsigned short* Wout_t  = (unsigned short*)alloc((size_t)NC * HH * 2);
    (void)ws_size; (void)in_sizes; (void)n_in; (void)out_size;

    const int TPB = 256;
    auto blocks = [](long long n) { return (unsigned)((n + 255) / 256); };

    // ---- input/weight conversion (bf16, weights K-major) ----
    k_cvt_bf16<<<blocks((long long)BL * DIN), TPB, 0, stream>>>(x, x_bf, BL * DIN);
    k_transpose_cvt<<<blocks((long long)DIN * HH), TPB, 0, stream>>>(W_in, Win_t, DIN, HH);
    k_transpose_cvt<<<blocks((long long)HH * 2 * DI), TPB, 0, stream>>>(in_proj_w, inp_t, HH, 2 * DI);
    k_transpose_cvt<<<blocks((long long)DI * (DTR + 2 * NN)), TPB, 0, stream>>>(x_proj_w, xproj_t, DI, DTR + 2 * NN);
    k_transpose_cvt<<<blocks((long long)DTR * DI), TPB, 0, stream>>>(dt_proj_w, dtp_t, DTR, DI);
    k_transpose_cvt<<<blocks((long long)DI * HH), TPB, 0, stream>>>(out_projw, outp_t, DI, HH);
    k_transpose_cvt<<<blocks((long long)HH * NC), TPB, 0, stream>>>(W_out, Wout_t, HH, NC);

    // N must be divisible by NT*16; choose widest exact tiling.
    auto gemm = [&](int M, int N, int Kt, const unsigned short* A,
                    const unsigned short* Wt, const float* bias,
                    float* Cf, unsigned short* Cb) {
        unsigned gy = (unsigned)((M + 127) / 128);
        if (N % 64 == 0) {
            dim3 grid(N / 64, gy);
            k_gemm_bf16<4><<<grid, 256, 0, stream>>>(M, N, Kt, A, Wt, bias, Cf, Cb);
        } else if (N % 32 == 0) {
            dim3 grid(N / 32, gy);
            k_gemm_bf16<2><<<grid, 256, 0, stream>>>(M, N, Kt, A, Wt, bias, Cf, Cb);
        } else {
            dim3 grid(N / 16, gy);
            k_gemm_bf16<1><<<grid, 256, 0, stream>>>(M, N, Kt, A, Wt, bias, Cf, Cb);
        }
    };

    // G1: h = x @ W_in + b_in           -> bf16 only (feeds G2)
    gemm(BL, HH, DIN, x_bf, Win_t, b_in, nullptr, h_bf);
    // G2: xz = h @ in_proj_w            -> f32 (xm for conv, z for gate)
    gemm(BL, 2 * DI, HH, h_bf, inp_t, nullptr, xz_f, nullptr);
    // conv + SiLU -> xc (f32 + bf16)
    k_conv_silu<<<blocks((long long)BL * DI), TPB, 0, stream>>>(xz_f, conv_w, conv_b, xc_f, xc_b);
    // G3: proj = xc @ x_proj_w (N=96 -> NT=2) -> f32
    gemm(BL, DTR + 2 * NN, DI, xc_b, xproj_t, nullptr, proj_f, nullptr);
    // dt slice -> bf16
    k_slice_dt<<<blocks((long long)BL * DTR), TPB, 0, stream>>>(proj_f, dt_bf);
    // G4: delta_raw = dt @ dt_proj_w + dt_proj_b -> f32 (softplus in scan)
    gemm(BL, DI, DTR, dt_bf, dtp_t, dt_proj_b, delta_f, nullptr);
    // selective scan + D*xc + SiLU(z) gate -> y bf16
    {
        dim3 grid(DI / 256, BB);
        k_scan<<<grid, 256, 0, stream>>>(delta_f, xc_f, proj_f, xz_f, A_log, Dp, y_bf);
    }
    // G5: out = y @ out_proj_w          -> bf16 only (feeds G6)
    gemm(BL, HH, DI, y_bf, outp_t, nullptr, nullptr, out_bf);
    // G6: logits = out @ W_out + b_out  -> f32 to d_out (N=16 -> NT=1)
    gemm(BL, NC, HH, out_bf, Wout_t, b_out, logits, nullptr);
}
